// HopfieldMemory_76141180223719
// MI455X (gfx1250) — compile-verified, matrix operand order checked by reading, not run on hardware
//
#include <hip/hip_runtime.h>

#define C_DIM 8192
#define M_DIM 4096
#define D_DIM 64
#define SLAB1 4
#define SLAB2 8
#define SLAB3 8

typedef __attribute__((ext_vector_type(16))) __bf16 v16bf;
typedef __attribute__((ext_vector_type(8)))  __bf16 v8bf;
typedef __attribute__((ext_vector_type(8)))  float  v8f;

// fp32 -> bf16, round-to-nearest-even
__device__ __forceinline__ __bf16 f2bf(float x) {
  unsigned u = __float_as_uint(x);
  unsigned r = (u + 0x7fffu + ((u >> 16) & 1u)) >> 16;
  unsigned short s = (unsigned short)r;
  return *(__bf16*)&s;
}

// 16-element bf16 fragment from two contiguous 16-byte chunks.
// ISA 16-bit A/B fragment layout: elements 0-7 = K kb+0..7,
// elements 8-15 = K kb+16..23 (kb = 8*(lane>>4)).
__device__ __forceinline__ v16bf mkfrag(const unsigned short* p0,
                                        const unsigned short* p1) {
  v8bf lo = *(const v8bf*)p0;
  v8bf hi = *(const v8bf*)p1;
  return __builtin_shufflevector(lo, hi, 0,1,2,3,4,5,6,7,8,9,10,11,12,13,14,15);
}

__device__ __forceinline__ v8f wmma_bf16(v16bf a, v16bf b, v8f c) {
  return __builtin_amdgcn_wmma_f32_16x16x32_bf16(false, a, false, b,
                                                 (short)0, c, false, false);
}

// Async global -> LDS copy of 16 bytes (ASYNCcnt-tracked; ISA §15.18.3 op 98).
// LDS offset = low 32 bits of the generic pointer (aperture is in the high
// half per ISA §10.2), so this is base-exact.
__device__ __forceinline__ void async_copy_b128(const unsigned short* lds_p,
                                                const unsigned short* g_p) {
  unsigned ldso = (unsigned)(uintptr_t)lds_p;
  asm volatile("global_load_async_to_lds_b128 %0, %1, off"
               :: "v"(ldso), "v"((unsigned long long)(uintptr_t)g_p)
               : "memory");
}
__device__ __forceinline__ void async_wait0() {
  asm volatile("s_wait_asynccnt 0x0" ::: "memory");
}

// ---------------------------------------------------------------------------
// Prep: Q,Km -> bf16 row-major; Q,O -> bf16 transposed (D x C) for stage-2 B.
// ---------------------------------------------------------------------------
__global__ __launch_bounds__(256) void k_prep(
    const float* __restrict__ Q, const float* __restrict__ O,
    const float* __restrict__ Km,
    unsigned short* __restrict__ Qb, unsigned short* __restrict__ Qt,
    unsigned short* __restrict__ Ot, unsigned short* __restrict__ Kb)
{
  int tid = blockIdx.x * 256 + threadIdx.x;
  if (tid < C_DIM * D_DIM) {
    float q = Q[tid];
    int c = tid >> 6, d = tid & 63;
    __bf16 qb = f2bf(q);
    Qb[tid] = *(unsigned short*)&qb;
    Qt[d * C_DIM + c] = *(unsigned short*)&qb;
    __bf16 ob = f2bf(O[tid]);
    Ot[d * C_DIM + c] = *(unsigned short*)&ob;
  }
  if (tid < M_DIM * D_DIM) {
    __bf16 kb = f2bf(Km[tid]);
    Kb[tid] = *(unsigned short*)&kb;
  }
}

// ---------------------------------------------------------------------------
// Pass 1: per-row (c) online softmax stats over an M-slab. All 8 waves of a
// block consume the SAME Km stream -> double-buffered async global->LDS
// pipeline shares it; per-lane online (max,sum) keeps the loop VALU-only.
// ---------------------------------------------------------------------------
__global__ __launch_bounds__(256) void k1_rowstats(
    const unsigned short* __restrict__ Qb, const unsigned short* __restrict__ Kb,
    float* __restrict__ pmax, float* __restrict__ psum)
{
  __shared__ __align__(16) unsigned short kbuf[2][16 * 64];  // 2 x 2KB

  const int tid   = threadIdx.x;
  const int wave  = blockIdx.x * 8 + (tid >> 5);
  const int lane  = tid & 31;
  const int ctile = wave % (C_DIM / 16);
  const int slab  = wave / (C_DIM / 16);   // block-uniform (512 waves/slab, 8/block)
  const int c0  = ctile * 16;
  const int row = lane & 15;
  const int kb  = (lane >> 4) << 3;
  const int ms   = M_DIM / SLAB1;
  const int mbeg = slab * ms, mend = mbeg + ms;

  // 128 threads copy one 16-row x 64-col bf16 Km tile (2KB) as 128 x b128.
  auto issue_tile = [&](int m0, int bi) {
    if (tid < 128) {
      int r = tid >> 3, s = tid & 7;
      async_copy_b128(&kbuf[bi][r * 64 + s * 8],
                      Kb + (size_t)(m0 + r) * D_DIM + s * 8);
    }
  };

  issue_tile(mbeg, 0);

  const unsigned short* qrow = Qb + (size_t)(c0 + row) * D_DIM;
  v16bf a0 = mkfrag(qrow + kb,      qrow + kb + 16);
  v16bf a1 = mkfrag(qrow + 32 + kb, qrow + 32 + kb + 16);

  float runMax[8], runSum[8];
#pragma unroll
  for (int r = 0; r < 8; ++r) { runMax[r] = -1e30f; runSum[r] = 0.f; }

  async_wait0();
  __syncthreads();

  for (int m0 = mbeg; m0 < mend; m0 += 16) {
    const int bi = ((m0 - mbeg) >> 4) & 1;
    if (m0 + 16 < mend) issue_tile(m0 + 16, bi ^ 1);  // overlap copy w/ compute

    const unsigned short* krow = &kbuf[bi][row * 64];
    v16bf b0 = mkfrag(krow + kb,      krow + kb + 16);
    v16bf b1 = mkfrag(krow + 32 + kb, krow + 32 + kb + 16);
    v8f acc = {};
    acc = wmma_bf16(a0, b0, acc);
    acc = wmma_bf16(a1, b1, acc);
#pragma unroll
    for (int r = 0; r < 8; ++r) {
      float v  = acc[r];
      float nm = fmaxf(runMax[r], v);
      runSum[r] = runSum[r] * __expf(runMax[r] - nm) + __expf(v - nm);
      runMax[r] = nm;
    }
    async_wait0();
    __syncthreads();
  }

  // One cross-lane (max,sum) pair-merge over the 16 lanes of each half.
#pragma unroll
  for (int r = 0; r < 8; ++r) {
    float m = runMax[r], s = runSum[r];
#pragma unroll
    for (int mask = 1; mask <= 8; mask <<= 1) {
      float mo = __shfl_xor(m, mask, 32);
      float so = __shfl_xor(s, mask, 32);
      float nm = fmaxf(m, mo);
      s = s * __expf(m - nm) + so * __expf(mo - nm);
      m = nm;
    }
    runMax[r] = m; runSum[r] = s;
  }
  if (row == 0) {
#pragma unroll
    for (int r = 0; r < 8; ++r) {
      int c = c0 + r + ((lane >> 4) << 3);
      pmax[slab * C_DIM + c] = runMax[r];
      psum[slab * C_DIM + c] = runSum[r];
    }
  }
}

// Merge per-slab (max,sum) partials -> rowmax, rowinv = 1/rowsum.
__global__ __launch_bounds__(256) void k1b_merge(
    const float* __restrict__ pmax, const float* __restrict__ psum,
    float* __restrict__ rowmax, float* __restrict__ rowinv)
{
  int c = blockIdx.x * 256 + threadIdx.x;
  float m = -1e30f, s = 0.f;
#pragma unroll
  for (int sl = 0; sl < SLAB1; ++sl) {
    float mo = pmax[sl * C_DIM + c];
    float so = psum[sl * C_DIM + c];
    float nm = fmaxf(m, mo);
    s = s * __expf(m - nm) + so * __expf(mo - nm);
    m = nm;
  }
  rowmax[c] = m;
  rowinv[c] = 1.0f / s;
}

// ---------------------------------------------------------------------------
// Pass 2: per-column (m) sum(exp(S1)) (Dc denominator; S1 in [0,1] so no max
// stabilization needed) and sum(S1) (alpha = 0.5 * mean_c S1).
// ---------------------------------------------------------------------------
__global__ __launch_bounds__(256) void k2_colstats(
    const unsigned short* __restrict__ Qb, const unsigned short* __restrict__ Kb,
    const float* __restrict__ rowmax, const float* __restrict__ rowinv,
    float* __restrict__ colsumexp, float* __restrict__ colsumS1)
{
  const int wave  = blockIdx.x * (blockDim.x >> 5) + (threadIdx.x >> 5);
  const int lane  = threadIdx.x & 31;
  const int mtile = wave % (M_DIM / 16);
  const int slab  = wave / (M_DIM / 16);
  const int m0  = mtile * 16;
  const int row = lane & 15;
  const int kb  = (lane >> 4) << 3;
  const int cs   = C_DIM / SLAB2;
  const int cbeg = slab * cs, cend = cbeg + cs;

  const unsigned short* krow = Kb + (size_t)(m0 + row) * D_DIM;
  v16bf b0 = mkfrag(krow + kb,      krow + kb + 16);
  v16bf b1 = mkfrag(krow + 32 + kb, krow + 32 + kb + 16);

  float sExp = 0.f, sS1 = 0.f;
  for (int c0 = cbeg; c0 < cend; c0 += 16) {
    const unsigned short* qrow = Qb + (size_t)(c0 + row) * D_DIM;
    v16bf a0 = mkfrag(qrow + kb,      qrow + kb + 16);
    v16bf a1 = mkfrag(qrow + 32 + kb, qrow + 32 + kb + 16);
    v8f acc = {};
    acc = wmma_bf16(a0, b0, acc);
    acc = wmma_bf16(a1, b1, acc);
    float rm[8], ri[8];
    *(float4*)&rm[0] = *(const float4*)(rowmax + c0 + kb);
    *(float4*)&rm[4] = *(const float4*)(rowmax + c0 + kb + 4);
    *(float4*)&ri[0] = *(const float4*)(rowinv + c0 + kb);
    *(float4*)&ri[4] = *(const float4*)(rowinv + c0 + kb + 4);
#pragma unroll
    for (int r = 0; r < 8; ++r) {
      float s1 = __expf(acc[r] - rm[r]) * ri[r];
      sExp += __expf(s1);
      sS1  += s1;
    }
  }
  sExp += __shfl_xor(sExp, 16, 32);
  sS1  += __shfl_xor(sS1, 16, 32);
  if (lane < 16) {
    atomicAdd(&colsumexp[m0 + lane], sExp);
    atomicAdd(&colsumS1[m0 + lane], sS1);
  }
}

// ---------------------------------------------------------------------------
// Pass 3: Km_target = Dc^T @ Q, Vm_target = Dc^T @ O. Block-shared Qb / Qt /
// Ot chunks are staged through a double-buffered async global->LDS pipeline;
// the WMMA C/D layout of the Am tiles feeds directly into the bf16 A-fragment
// of the stage-2 WMMA after the exp transform.
// ---------------------------------------------------------------------------
__global__ __launch_bounds__(256) void k3_targets(
    const unsigned short* __restrict__ Qb, const unsigned short* __restrict__ Kb,
    const unsigned short* __restrict__ Qt, const unsigned short* __restrict__ Ot,
    const float* __restrict__ rowmax, const float* __restrict__ rowinv,
    const float* __restrict__ colsumexp,
    float* __restrict__ Ktgt, float* __restrict__ Vtgt)
{
  __shared__ __align__(16) unsigned short qbuf[2][32 * 64];   // 2 x 4KB
  __shared__ __align__(16) unsigned short qtbuf[2][64 * 32];  // 2 x 4KB
  __shared__ __align__(16) unsigned short otbuf[2][64 * 32];  // 2 x 4KB

  const int tid   = threadIdx.x;
  const int wave  = blockIdx.x * 8 + (tid >> 5);
  const int lane  = tid & 31;
  const int mtile = wave % (M_DIM / 16);
  const int slab  = wave / (M_DIM / 16);   // block-uniform (256 waves/slab)
  const int m0  = mtile * 16;
  const int row = lane & 15;
  const int kb  = (lane >> 4) << 3;
  const int cs   = C_DIM / SLAB3;
  const int cbeg = slab * cs, cend = cbeg + cs;

  // 256 threads copy a 32-c chunk: Qb rows (4KB) + Qt cols (4KB) + Ot (4KB).
  auto issue_chunk = [&](int c0, int bi) {
    {
      int r = tid >> 3, s = tid & 7;   // Qb: 32 rows x 8 segs
      async_copy_b128(&qbuf[bi][r * 64 + s * 8],
                      Qb + (size_t)(c0 + r) * D_DIM + s * 8);
    }
    {
      int d = tid >> 2, s = tid & 3;   // Qt/Ot: 64 rows x 4 segs of the chunk
      async_copy_b128(&qtbuf[bi][d * 32 + s * 8],
                      Qt + (size_t)d * C_DIM + c0 + s * 8);
      async_copy_b128(&otbuf[bi][d * 32 + s * 8],
                      Ot + (size_t)d * C_DIM + c0 + s * 8);
    }
  };

  issue_chunk(cbeg, 0);

  const unsigned short* krow = Kb + (size_t)(m0 + row) * D_DIM;
  v16bf kf0 = mkfrag(krow + kb,      krow + kb + 16);
  v16bf kf1 = mkfrag(krow + 32 + kb, krow + 32 + kb + 16);

  const float invDen = 1.0f / colsumexp[m0 + row];

  v8f accK[4] = {};
  v8f accV[4] = {};

  async_wait0();
  __syncthreads();

  for (int c0 = cbeg; c0 < cend; c0 += 32) {
    const int bi = ((c0 - cbeg) >> 5) & 1;
    if (c0 + 32 < cend) issue_chunk(c0 + 32, bi ^ 1);  // overlap with compute

    // Stage 1: two 16(c) x 16(m) Am tiles via WMMA (K over D = 64 in 2 steps)
    const unsigned short* q0 = &qbuf[bi][row * 64];
    v16bf a00 = mkfrag(q0 + kb,      q0 + kb + 16);
    v16bf a01 = mkfrag(q0 + 32 + kb, q0 + 32 + kb + 16);
    v8f t0 = {};
    t0 = wmma_bf16(a00, kf0, t0);
    t0 = wmma_bf16(a01, kf1, t0);

    const unsigned short* q1 = &qbuf[bi][(16 + row) * 64];
    v16bf a10 = mkfrag(q1 + kb,      q1 + kb + 16);
    v16bf a11 = mkfrag(q1 + 32 + kb, q1 + 32 + kb + 16);
    v8f t1 = {};
    t1 = wmma_bf16(a10, kf0, t1);
    t1 = wmma_bf16(a11, kf1, t1);

    // Row stats (contiguous float4 loads) + exp transform -> bf16 A-fragment
    float rmA[8], riA[8], rmB[8], riB[8];
    *(float4*)&rmA[0] = *(const float4*)(rowmax + c0 + kb);
    *(float4*)&rmA[4] = *(const float4*)(rowmax + c0 + kb + 4);
    *(float4*)&riA[0] = *(const float4*)(rowinv + c0 + kb);
    *(float4*)&riA[4] = *(const float4*)(rowinv + c0 + kb + 4);
    *(float4*)&rmB[0] = *(const float4*)(rowmax + c0 + 16 + kb);
    *(float4*)&rmB[4] = *(const float4*)(rowmax + c0 + 16 + kb + 4);
    *(float4*)&riB[0] = *(const float4*)(rowinv + c0 + 16 + kb);
    *(float4*)&riB[4] = *(const float4*)(rowinv + c0 + 16 + kb + 4);

    v16bf dc;
#pragma unroll
    for (int i = 0; i < 8; ++i) {
      float s1a = __expf(t0[i] - rmA[i]) * riA[i];
      dc[i] = f2bf(__expf(s1a) * invDen);
      float s1b = __expf(t1[i] - rmB[i]) * riB[i];
      dc[i + 8] = f2bf(__expf(s1b) * invDen);
    }

    // Stage 2: accumulate Dc^T @ Q and Dc^T @ O (B fragments from LDS)
#pragma unroll
    for (int t = 0; t < 4; ++t) {
      const unsigned short* qt = &qtbuf[bi][(t * 16 + row) * 32];
      v16bf bq = mkfrag(qt + kb, qt + kb + 16);
      accK[t] = wmma_bf16(dc, bq, accK[t]);
      const unsigned short* ot = &otbuf[bi][(t * 16 + row) * 32];
      v16bf bo = mkfrag(ot + kb, ot + kb + 16);
      accV[t] = wmma_bf16(dc, bo, accV[t]);
    }

    async_wait0();
    __syncthreads();
  }

#pragma unroll
  for (int t = 0; t < 4; ++t) {
#pragma unroll
    for (int r = 0; r < 8; ++r) {
      int m = m0 + r + ((lane >> 4) << 3);
      int d = t * 16 + row;
      atomicAdd(&Ktgt[m * D_DIM + d], accK[t][r]);
      atomicAdd(&Vtgt[m * D_DIM + d], accV[t][r]);
    }
  }
}

// ---------------------------------------------------------------------------
// Finalize: gKm = alpha * (Km_target - Km), gVm = alpha * (Vm_target - Vm)
// alpha[m] = 0.5 * colsumS1[m] / C   (smooth_max at temperature 0 == mean)
// ---------------------------------------------------------------------------
__global__ __launch_bounds__(256) void k4_final(
    const float* __restrict__ Km, const float* __restrict__ Vm,
    const float* __restrict__ Ktgt, const float* __restrict__ Vtgt,
    const float* __restrict__ colsumS1, float* __restrict__ out)
{
  int tid = blockIdx.x * 256 + threadIdx.x;
  int m = tid >> 6;
  float alpha = 0.5f * colsumS1[m] * (1.0f / (float)C_DIM);
  out[tid] = alpha * (Ktgt[tid] - Km[tid]);
  out[M_DIM * D_DIM + tid] = alpha * (Vtgt[tid] - Vm[tid]);
}

extern "C" void kernel_launch(void* const* d_in, const int* in_sizes, int n_in,
                              void* d_out, int out_size, void* d_ws, size_t ws_size,
                              hipStream_t stream)
{
  (void)in_sizes; (void)n_in; (void)out_size; (void)ws_size;
  const float* Q  = (const float*)d_in[0];
  const float* O  = (const float*)d_in[1];
  const float* Km = (const float*)d_in[2];
  const float* Vm = (const float*)d_in[3];
  float* out = (float*)d_out;

  char* ws = (char*)d_ws;
  size_t off = 0;
  unsigned short* Qb = (unsigned short*)(ws + off); off += (size_t)C_DIM * D_DIM * 2;
  unsigned short* Qt = (unsigned short*)(ws + off); off += (size_t)C_DIM * D_DIM * 2;
  unsigned short* Ot = (unsigned short*)(ws + off); off += (size_t)C_DIM * D_DIM * 2;
  unsigned short* Kb = (unsigned short*)(ws + off); off += (size_t)M_DIM * D_DIM * 2;
  float* rowmax = (float*)(ws + off); off += (size_t)C_DIM * 4;
  float* rowinv = (float*)(ws + off); off += (size_t)C_DIM * 4;
  float* pmax = (float*)(ws + off); off += (size_t)SLAB1 * C_DIM * 4;
  float* psum = (float*)(ws + off); off += (size_t)SLAB1 * C_DIM * 4;
  size_t zbeg = off;
  float* colsumexp = (float*)(ws + off); off += (size_t)M_DIM * 4;
  float* colsumS1  = (float*)(ws + off); off += (size_t)M_DIM * 4;
  float* Ktgt = (float*)(ws + off); off += (size_t)M_DIM * D_DIM * 4;
  float* Vtgt = (float*)(ws + off); off += (size_t)M_DIM * D_DIM * 4;
  size_t zend = off;

  hipMemsetAsync(ws + zbeg, 0, zend - zbeg, stream);  // zero accumulators

  k_prep<<<(C_DIM * D_DIM + 255) / 256, 256, 0, stream>>>(Q, O, Km, Qb, Qt, Ot, Kb);
  k1_rowstats<<<SLAB1 * (C_DIM / 16) / 8, 256, 0, stream>>>(Qb, Kb, pmax, psum);
  k1b_merge<<<C_DIM / 256, 256, 0, stream>>>(pmax, psum, rowmax, rowinv);
  k2_colstats<<<(M_DIM / 16) * SLAB2 / 8, 256, 0, stream>>>(Qb, Kb, rowmax, rowinv,
                                                            colsumexp, colsumS1);
  k3_targets<<<(M_DIM / 16) * SLAB3 / 8, 256, 0, stream>>>(Qb, Kb, Qt, Ot, rowmax, rowinv,
                                                           colsumexp, Ktgt, Vtgt);
  k4_final<<<(M_DIM * D_DIM) / 256, 256, 0, stream>>>(Km, Vm, Ktgt, Vtgt, colsumS1, out);
}
